// MultiHeadAttention_27908697490220
// MI455X (gfx1250) — compile-verified, matrix operand order checked by reading, not run on hardware
//
#include <hip/hip_runtime.h>
#include <hip/hip_bf16.h>
#include <math.h>
#include <stdint.h>

// MI455X / gfx1250 multi-head attention, bf16 WMMA + async-LDS staging.
// B=4, S=2048, D=1024, H=16, dk=64.

#define D_MODEL 1024
#define NHEADS  16
#define DKH     64
#define SEQ     2048
#define BATCH   4

typedef __attribute__((ext_vector_type(16))) __bf16 v16bf;
typedef __attribute__((ext_vector_type(8)))  float  v8f;

#define WMMA_BF16(A, B, C) \
  __builtin_amdgcn_wmma_f32_16x16x32_bf16(false, (A), false, (B), (short)0, (C), false, false)

// ---- CDNA5 async global->LDS copy (ASYNCcnt-tracked), inline asm ----------
static __device__ __forceinline__ void async_ld_b128(uint32_t lds_off, const void* gaddr) {
  asm volatile("global_load_async_to_lds_b128 %0, %1, off"
               :: "v"(lds_off), "v"(gaddr) : "memory");
}
template <int N>
static __device__ __forceinline__ void wait_async() {
  asm volatile("s_wait_asynccnt %0" :: "i"(N) : "memory");
}

// ---- A-fragment loaders (16x32 bf16, ISA 7.12.2 layout) -------------------
// lane<16: row M=lane,    K in {kb..kb+7, kb+16..kb+23} with kb=0
// lane>=16: row M=lane-16, same pattern with kb=8
static __device__ __forceinline__ v16bf load_a_f32(const float* base, int ld, int lane) {
  const int row = lane & 15;
  const int kb  = (lane >> 4) << 3;
  const float* p = base + (size_t)row * ld + kb;
  v16bf a;
#pragma unroll
  for (int i = 0; i < 8; ++i) {
    a[i]     = (__bf16)p[i];        // K = kb + i
    a[i + 8] = (__bf16)p[i + 16];   // K = kb + 16 + i
  }
  return a;
}

static __device__ __forceinline__ v16bf load_a_bf16(const __bf16* base, int ld, int lane) {
  const int row = lane & 15;
  const int kb  = (lane >> 4) << 3;
  const __bf16* p = base + (size_t)row * ld + kb;
  v16bf a;
#pragma unroll
  for (int i = 0; i < 8; ++i) {
    a[i]     = p[i];
    a[i + 8] = p[i + 16];
  }
  return a;
}

// ---------------------------------------------------------------------------
// Projection: Y = X @ W^T + b   (X fp32 [M,1024], W fp32 [1024,1024] row=out)
// mode 0: Y stored bf16 [M, D_MODEL]
// mode 1: Y stored bf16 transposed per head: Vt[b, h, d, s]  (for PV matmul)
// ---------------------------------------------------------------------------
__global__ __launch_bounds__(256) void proj_kernel(
    const float* __restrict__ X, const float* __restrict__ W,
    const float* __restrict__ bias, __bf16* __restrict__ Y, int mode) {
  const int tid = threadIdx.x, wave = tid >> 5, lane = tid & 31;
  const int m0 = blockIdx.x * 32 + (wave >> 2) * 16;
  const int n0 = blockIdx.y * 64 + (wave & 3) * 16;
  const float* Xb = X + (size_t)m0 * D_MODEL;
  const float* Wb = W + (size_t)n0 * D_MODEL;  // row n of W == col n of W^T
  v8f acc = {};
  for (int kk = 0; kk < D_MODEL; kk += 32) {
    if (kk + 32 < D_MODEL) {  // cover next tile: global_prefetch_b8
      __builtin_prefetch(Xb + (size_t)(lane & 15) * D_MODEL + kk + 32, 0, 1);
      __builtin_prefetch(Wb + (size_t)(lane & 15) * D_MODEL + kk + 32, 0, 1);
    }
    v16bf a = load_a_f32(Xb + kk, D_MODEL, lane);
    v16bf b = load_a_f32(Wb + kk, D_MODEL, lane);
    acc = WMMA_BF16(a, b, acc);
  }
  const int n  = lane & 15;
  const int mh = (lane >> 4) << 3;
  const float bv = bias[n0 + n];
  if (mode == 0) {
#pragma unroll
    for (int r = 0; r < 8; ++r)
      Y[(size_t)(m0 + mh + r) * D_MODEL + n0 + n] = (__bf16)(acc[r] + bv);
  } else {
    const int nn = n0 + n, h = nn >> 6, d = nn & 63;
#pragma unroll
    for (int r = 0; r < 8; ++r) {
      const int m = m0 + mh + r, b = m >> 11, s = m & (SEQ - 1);
      Y[(((size_t)(b * NHEADS + h) * DKH) + d) * SEQ + s] = (__bf16)(acc[r] + bv);
    }
  }
}

// ---------------------------------------------------------------------------
// Attention core: one WG per (b, h, 16-query-row block).
// Scores/probs (16 x 2048 bf16 = 64 KB) + per-wave double-buffered K-tile
// staging (8 x 2 x 2 KB = 32 KB) filled by global_load_async_to_lds_b128.
// ---------------------------------------------------------------------------
__global__ __launch_bounds__(256) void attn_kernel(
    const __bf16* __restrict__ Qb,   // [B,S,D] bf16
    const __bf16* __restrict__ Kb,   // [B,S,D] bf16
    const __bf16* __restrict__ Vt,   // [B,H,DKH,SEQ] bf16 (transposed)
    float* __restrict__ attn_out,    // [B,H,S,S] fp32 (output)
    __bf16* __restrict__ Ctx)        // [B,S,D] bf16
{
  __shared__ __bf16 sc[16 * SEQ];          // 64 KB scores -> probabilities
  __shared__ __bf16 kst[8][2][16 * DKH];   // 32 KB K-tile staging
  __shared__ float rmax[16][16];
  __shared__ float rsum[16][16];

  const int wg = blockIdx.x;
  const int qt = wg & (SEQ / 16 - 1);          // query tile 0..127
  const int bh = wg / (SEQ / 16);
  const int h  = bh & (NHEADS - 1);
  const int b  = bh / NHEADS;
  const int tid = threadIdx.x, wave = tid >> 5, lane = tid & 31;

  const __bf16* Qbase = Qb + ((size_t)b * SEQ + qt * 16) * D_MODEL + h * DKH;
  const __bf16* Kbase = Kb + (size_t)b * SEQ * D_MODEL + h * DKH;

  // Q fragments for this 16-row block (K = 0..31, 32..63)
  const v16bf qf0 = load_a_bf16(Qbase,      D_MODEL, lane);
  const v16bf qf1 = load_a_bf16(Qbase + 32, D_MODEL, lane);

  // Async staging: 16 rows x 128 B per K tile; lane covers 64 B (4 x b128).
  const uint32_t kst_base = (uint32_t)(uintptr_t)(&kst[wave][0][0]);
  const int srow = lane & 15;
  const int c0   = (lane >> 4) * 4;              // chunk base (16 B chunks)
  const uint32_t lrow = (uint32_t)(srow * 128 + c0 * 16);

  // ---- scores = (Q K^T) * 1/sqrt(dk); each wave covers 16 key tiles -------
  {
    // prologue: issue tile t=0 into buffer 0
    const __bf16* Kt0 = Kbase + (size_t)wave * 16 * D_MODEL +
                        (size_t)srow * D_MODEL + c0 * 8;
#pragma unroll
    for (int c = 0; c < 4; ++c)
      async_ld_b128(kst_base + lrow + c * 16, Kt0 + c * 8);
  }
  for (int t = 0; t < 16; ++t) {
    const int j = wave + t * 8;                  // key tile index
    if (t + 1 < 16) {
      const __bf16* Ktn = Kbase + (size_t)(wave + (t + 1) * 8) * 16 * D_MODEL +
                          (size_t)srow * D_MODEL + c0 * 8;
      const uint32_t buf = (uint32_t)(((t + 1) & 1) * (16 * DKH * 2));
#pragma unroll
      for (int c = 0; c < 4; ++c)
        async_ld_b128(kst_base + buf + lrow + c * 16, Ktn + c * 8);
      wait_async<4>();   // allow the 4 just-issued copies to remain in flight
    } else {
      wait_async<0>();
    }
    const __bf16* kbuf = &kst[wave][t & 1][0];
    v16bf kf0 = load_a_bf16(kbuf,      DKH, lane);
    v16bf kf1 = load_a_bf16(kbuf + 32, DKH, lane);
    v8f acc = {};
    acc = WMMA_BF16(qf0, kf0, acc);
    acc = WMMA_BF16(qf1, kf1, acc);
    const int n  = lane & 15;
    const int mh = (lane >> 4) << 3;
#pragma unroll
    for (int r = 0; r < 8; ++r)
      sc[(mh + r) * SEQ + j * 16 + n] = (__bf16)(acc[r] * 0.125f);  // 1/sqrt(64)
  }
  __syncthreads();

  // ---- softmax over each of the 16 rows (16 threads per row, coalesced) ---
  const int row  = tid >> 4;     // 0..15
  const int slot = tid & 15;     // 0..15
  float m = -3.0e38f;
  for (int i = slot; i < SEQ; i += 16) m = fmaxf(m, (float)sc[row * SEQ + i]);
  rmax[row][slot] = m;
  __syncthreads();
#pragma unroll
  for (int t = 0; t < 16; ++t) m = fmaxf(m, rmax[row][t]);
  float s = 0.0f;
  for (int i = slot; i < SEQ; i += 16) s += __expf((float)sc[row * SEQ + i] - m);
  rsum[row][slot] = s;
  __syncthreads();
  s = 0.0f;
#pragma unroll
  for (int t = 0; t < 16; ++t) s += rsum[row][t];
  const float inv = 1.0f / s;

  float* arow = attn_out + (size_t)((b * NHEADS + h) * SEQ + qt * 16 + row) * SEQ;
  for (int i = slot; i < SEQ; i += 16) {
    const float p = __expf((float)sc[row * SEQ + i] - m) * inv;
    sc[row * SEQ + i] = (__bf16)p;  // keep bf16 probs for PV matmul
    arow[i] = p;                    // fp32 attn output (1 GiB; bandwidth floor)
  }
  __syncthreads();

  // ---- ctx = P @ V : waves 0..3 each produce a 16x16 dk tile over K=2048 --
  if (wave < 4) {
    const __bf16* Vb = Vt + ((size_t)(b * NHEADS + h) * DKH + wave * 16) * SEQ;
    v8f acc = {};
    for (int kk = 0; kk < SEQ; kk += 32) {
      v16bf a   = load_a_bf16(&sc[0] + kk, SEQ, lane);  // probs, row=q
      v16bf bfr = load_a_bf16(Vb + kk, SEQ, lane);      // rows = dk, K = keys
      acc = WMMA_BF16(a, bfr, acc);
    }
    const int n  = lane & 15;
    const int mh = (lane >> 4) << 3;
#pragma unroll
    for (int r = 0; r < 8; ++r)
      Ctx[((size_t)b * SEQ + qt * 16 + mh + r) * D_MODEL + h * DKH + wave * 16 + n] =
          (__bf16)acc[r];
  }
}

// ---------------------------------------------------------------------------
// Output projection: out = Ctx(bf16) @ W_o^T + b_o  -> fp32
// ---------------------------------------------------------------------------
__global__ __launch_bounds__(256) void outproj_kernel(
    const __bf16* __restrict__ X, const float* __restrict__ W,
    const float* __restrict__ bias, float* __restrict__ Y) {
  const int tid = threadIdx.x, wave = tid >> 5, lane = tid & 31;
  const int m0 = blockIdx.x * 32 + (wave >> 2) * 16;
  const int n0 = blockIdx.y * 64 + (wave & 3) * 16;
  const __bf16* Xb = X + (size_t)m0 * D_MODEL;
  const float*  Wb = W + (size_t)n0 * D_MODEL;
  v8f acc = {};
  for (int kk = 0; kk < D_MODEL; kk += 32) {
    v16bf a = load_a_bf16(Xb + kk, D_MODEL, lane);
    v16bf b = load_a_f32 (Wb + kk, D_MODEL, lane);
    acc = WMMA_BF16(a, b, acc);
  }
  const int n  = lane & 15;
  const int mh = (lane >> 4) << 3;
  const float bv = bias[n0 + n];
#pragma unroll
  for (int r = 0; r < 8; ++r)
    Y[(size_t)(m0 + mh + r) * D_MODEL + n0 + n] = acc[r] + bv;
}

// ---------------------------------------------------------------------------
extern "C" void kernel_launch(void* const* d_in, const int* in_sizes, int n_in,
                              void* d_out, int out_size, void* d_ws, size_t ws_size,
                              hipStream_t stream) {
  const float* q  = (const float*)d_in[0];
  const float* k  = (const float*)d_in[1];
  const float* v  = (const float*)d_in[2];
  const float* Wq = (const float*)d_in[3];
  const float* bq = (const float*)d_in[4];
  const float* Wk = (const float*)d_in[5];
  const float* bk = (const float*)d_in[6];
  const float* Wv = (const float*)d_in[7];
  const float* bv = (const float*)d_in[8];
  const float* Wo = (const float*)d_in[9];
  const float* bo = (const float*)d_in[10];

  const size_t elems = (size_t)BATCH * SEQ * D_MODEL;  // 8,388,608

  // Workspace: 4 x bf16 [B,S,D] = 64 MB total
  __bf16* Qb  = (__bf16*)d_ws;
  __bf16* Kb  = Qb + elems;
  __bf16* Vt  = Kb + elems;
  __bf16* Ctx = Vt + elems;

  float* out_ptr  = (float*)d_out;           // [B,S,D] fp32
  float* attn_ptr = out_ptr + elems;         // [B,H,S,S] fp32

  dim3 pgrid(BATCH * SEQ / 32, D_MODEL / 64);
  proj_kernel<<<pgrid, 256, 0, stream>>>(q, Wq, bq, Qb, 0);
  proj_kernel<<<pgrid, 256, 0, stream>>>(k, Wk, bk, Kb, 0);
  proj_kernel<<<pgrid, 256, 0, stream>>>(v, Wv, bv, Vt, 1);

  attn_kernel<<<BATCH * NHEADS * (SEQ / 16), 256, 0, stream>>>(Qb, Kb, Vt, attn_ptr, Ctx);

  outproj_kernel<<<pgrid, 256, 0, stream>>>(Ctx, Wo, bo, out_ptr);
}
